// Transformer_77644418777357
// MI455X (gfx1250) — matrix operand order repaired and checked by reading, not compiled, not run on hardware
//
#include <hip/hip_runtime.h>
#include <hip/hip_bf16.h>

// ---------------------------------------------------------------------------
// Transformer block for MI455X (gfx1250, wave32, WMMA bf16 -> f32 accum)
// B=2, S=2048, D_MODEL=1024, H=16, D_HEAD=64, D_FF=4096
// ---------------------------------------------------------------------------

typedef __attribute__((ext_vector_type(16))) __bf16 v16bf;
typedef __attribute__((ext_vector_type(8)))  __bf16 v8bf;
typedef __attribute__((ext_vector_type(8)))  float  v8f;

union FragU { v16bf v; v8bf h[2]; };

// A-matrix fragment (16x32 bf16, M x K). Per CDNA5 ISA:
// lane L: row M = L&15 ; K-half offset = (L>>4)*8 ; VGPR0..3 = K[half..half+7],
// VGPR4..7 = K[16+half..16+half+7]  -> two contiguous 16B loads per lane.
__device__ __forceinline__ v16bf load_fragA(const __bf16* base, int ld, int rowBase, int kBase) {
  int lane = threadIdx.x & 31;
  const __bf16* p = base + (size_t)(rowBase + (lane & 15)) * (size_t)ld
                         + kBase + ((lane >> 4) << 3);
  FragU f;
  f.h[0] = *(const v8bf*)p;
  f.h[1] = *(const v8bf*)(p + 16);
  return f.v;
}

// B-matrix fragment (32x16 bf16, K x N), source given "transposed" as [N, K]
// row-major (exactly how W[out,in] and K/V per-head data are stored).
// lane L holds column N = L&15 with K = (L>>4)*8 .. +7 and 16+(L>>4)*8 .. +7.
// Optional prefetch of the next K-step (streamed weights).
template <bool PF>
__device__ __forceinline__ v16bf load_fragB(const __bf16* base, int ld, int rowBase, int kBase) {
  int lane = threadIdx.x & 31;
  const __bf16* p = base + (size_t)(rowBase + (lane & 15)) * (size_t)ld
                         + kBase + ((lane >> 4) << 3);
  FragU f;
  f.h[0] = *(const v8bf*)p;
  f.h[1] = *(const v8bf*)(p + 16);
  if (PF) __builtin_prefetch(p + 32, 0, 1);  // next K-step, same row
  return f.v;
}

__device__ __forceinline__ v8f wmma_bf16(v16bf a, v16bf b, v8f c) {
  return __builtin_amdgcn_wmma_f32_16x16x32_bf16(false, a, false, b, (short)0, c, false, false);
}

// ---------------------------------------------------------------------------
// f32 -> bf16 cast
// ---------------------------------------------------------------------------
__global__ void cast_f32_bf16(const float* __restrict__ src, __bf16* __restrict__ dst, int n) {
  int i = blockIdx.x * blockDim.x + threadIdx.x;
  if (i < n) dst[i] = (__bf16)src[i];
}

// ---------------------------------------------------------------------------
// Generic WMMA GEMM: out[m,n] = sum_k A[m,k] * W[n,k] + bias[n]  (+ epilogue)
// One wave computes a 32(M) x 64(N) tile: 2 A-frags x 4 B-frags = 8 WMMAs per
// K-step. 256 threads = 8 waves per block; waves of a block share the M-tile
// so A fragments hit WGP$.
// ---------------------------------------------------------------------------
enum { MODE_QPROJ = 0, MODE_KPROJ = 1, MODE_VPROJ = 2, MODE_FF1 = 3, MODE_FF2 = 4 };

template <int MODE>
__global__ __launch_bounds__(256) void gemm_wmma(
    const __bf16* __restrict__ A, const __bf16* __restrict__ W,
    const float* __restrict__ bias, const float* __restrict__ residual,
    float* __restrict__ outF, __bf16* __restrict__ outB,
    int M, int N, int K) {
  int w  = blockIdx.x * (blockDim.x >> 5) + (threadIdx.x >> 5);
  int nT = N >> 6;
  int mT = M >> 5;
  if (w >= mT * nT) return;  // wave-uniform
  int m0 = (w / nT) << 5;
  int n0 = (w % nT) << 6;
  int lane = threadIdx.x & 31;

  v8f acc0[4] = {};
  v8f acc1[4] = {};
  for (int k0 = 0; k0 < K; k0 += 32) {
    v16bf a0 = load_fragA(A, K, m0, k0);
    v16bf a1 = load_fragA(A, K, m0 + 16, k0);
#pragma unroll
    for (int t = 0; t < 4; ++t) {
      v16bf b = load_fragB<true>(W, K, n0 + (t << 4), k0);
      acc0[t] = wmma_bf16(a0, b, acc0[t]);
      acc1[t] = wmma_bf16(a1, b, acc1[t]);
    }
  }

  int colLane = lane & 15;
  int rowHalf = (lane >> 4) << 3;  // D-frag: lane holds col n, rows m = rowHalf + r
#pragma unroll
  for (int mi = 0; mi < 2; ++mi) {
#pragma unroll
    for (int t = 0; t < 4; ++t) {
      int n = n0 + (t << 4) + colLane;
      float bi = bias[n];
#pragma unroll
      for (int r = 0; r < 8; ++r) {
        int m = m0 + (mi << 4) + rowHalf + r;
        float val = (mi ? acc1[t][r] : acc0[t][r]) + bi;
        if (MODE == MODE_QPROJ) {
          outB[(size_t)m * N + n] = (__bf16)(val * 0.125f);  // fold 1/sqrt(64)
        } else if (MODE == MODE_KPROJ) {
          outB[(size_t)m * N + n] = (__bf16)val;
        } else if (MODE == MODE_VPROJ) {
          // store head-transposed: Vt[((b*16+h)*64+d)*2048 + s]
          int b_ = m >> 11, s = m & 2047;
          int h  = n >> 6,  d = n & 63;
          outB[((size_t)((b_ * 16 + h) << 6) + d) * 2048 + s] = (__bf16)val;
        } else if (MODE == MODE_FF1) {
          float gl = 0.5f * val * (1.0f + erff(val * 0.70710678118654752f));  // exact GELU
          outB[(size_t)m * N + n] = (__bf16)gl;
        } else {  // MODE_FF2: final output = GEMM + bias + att_ln residual
          outF[(size_t)m * N + n] = val + residual[(size_t)m * N + n];
        }
      }
    }
  }
}

// ---------------------------------------------------------------------------
// Flash attention: one wave per (batch, head, 16-row q-tile).
// Scores via WMMA (Q bf16 prescaled), online softmax in f32 registers,
// P staged through per-wave LDS (D-frag -> A-frag relayout), P@V via WMMA
// against head-transposed V. Writes att + q (residual fused).
// ---------------------------------------------------------------------------
__global__ __launch_bounds__(256) void attn_flash(
    const __bf16* __restrict__ Qb, const __bf16* __restrict__ Kb,
    const __bf16* __restrict__ Vt, const float* __restrict__ qres,
    float* __restrict__ outAttH, const int* __restrict__ causalFlag) {
  const int S = 2048, DM = 1024, H = 16, QT = S / 16;  // 128 q-tiles
  int w    = blockIdx.x * (blockDim.x >> 5) + (threadIdx.x >> 5);
  int qt   = w % QT;
  int h    = (w / QT) % H;
  int b    = w / (QT * H);
  int lane = threadIdx.x & 31;
  int wib  = threadIdx.x >> 5;
  int causal = *causalFlag;

  __shared__ __align__(16) __bf16 plds[8][16 * 32];  // per-wave 16x32 P tile

  int rowQ = b * S + qt * 16;
  v16bf aq0 = load_fragA(Qb, DM, rowQ, h * 64);
  v16bf aq1 = load_fragA(Qb, DM, rowQ, h * 64 + 32);

  v8f accO[4] = {};
  float mrun[8], lrun[8];
#pragma unroll
  for (int r = 0; r < 8; ++r) { mrun[r] = -3.0e38f; lrun[r] = 0.0f; }

  int colLane = lane & 15;
  int rowHalf = (lane >> 4) << 3;
  int iBase   = qt * 16 + rowHalf;  // query row index for register r: iBase + r

  int kend = causal ? ((qt * 16 + 16 + 31) & ~31) : S;
  if (kend > S) kend = S;

  for (int kc = 0; kc < kend; kc += 32) {
    // ---- scores: two 16x16 tiles over K-dim d=64 (2 chained WMMAs each) ----
    v8f sc0 = {}, sc1 = {};
    {
      v16bf bk = load_fragB<false>(Kb, DM, b * S + kc, h * 64);
      sc0 = wmma_bf16(aq0, bk, sc0);
      bk = load_fragB<false>(Kb, DM, b * S + kc, h * 64 + 32);
      sc0 = wmma_bf16(aq1, bk, sc0);
      bk = load_fragB<false>(Kb, DM, b * S + kc + 16, h * 64);
      sc1 = wmma_bf16(aq0, bk, sc1);
      bk = load_fragB<false>(Kb, DM, b * S + kc + 16, h * 64 + 32);
      sc1 = wmma_bf16(aq1, bk, sc1);
    }

    // ---- causal mask + online softmax (16-lane-half reductions) ----
    int j0 = kc + colLane, j1 = kc + 16 + colLane;
    float corr[8];
#pragma unroll
    for (int r = 0; r < 8; ++r) {
      float s0 = sc0[r], s1 = sc1[r];
      if (causal) {
        int i = iBase + r;
        if (j0 > i) s0 = -1.0e30f;
        if (j1 > i) s1 = -1.0e30f;
      }
      float mc = fmaxf(s0, s1);
      mc = fmaxf(mc, __shfl_xor(mc, 1));
      mc = fmaxf(mc, __shfl_xor(mc, 2));
      mc = fmaxf(mc, __shfl_xor(mc, 4));
      mc = fmaxf(mc, __shfl_xor(mc, 8));
      float mnew = fmaxf(mrun[r], mc);
      float p0 = __expf(s0 - mnew);
      float p1 = __expf(s1 - mnew);
      float rs = p0 + p1;
      rs += __shfl_xor(rs, 1);
      rs += __shfl_xor(rs, 2);
      rs += __shfl_xor(rs, 4);
      rs += __shfl_xor(rs, 8);
      corr[r] = __expf(mrun[r] - mnew);
      lrun[r] = lrun[r] * corr[r] + rs;
      mrun[r] = mnew;
      sc0[r] = p0; sc1[r] = p1;
    }
#pragma unroll
    for (int t = 0; t < 4; ++t)
#pragma unroll
      for (int r = 0; r < 8; ++r) accO[t][r] *= corr[r];

    // ---- stage P (bf16) to LDS in row-major, reload as A-fragment ----
#pragma unroll
    for (int r = 0; r < 8; ++r) {
      int m = rowHalf + r;
      plds[wib][m * 32 + colLane]      = (__bf16)sc0[r];
      plds[wib][m * 32 + 16 + colLane] = (__bf16)sc1[r];
    }
    // single-wave data: LDS ops are in-order; compiler inserts dscnt waits.
    FragU pf;
    const __bf16* pp = &plds[wib][(lane & 15) * 32 + ((lane >> 4) << 3)];
    pf.h[0] = *(const v8bf*)pp;
    pf.h[1] = *(const v8bf*)(pp + 16);

    // ---- accO += P @ V   (V head-transposed -> contiguous B fragments) ----
#pragma unroll
    for (int t = 0; t < 4; ++t) {
      v16bf bv = load_fragB<false>(Vt, S, (b * H + h) * 64 + (t << 4), kc);
      accO[t] = wmma_bf16(pf.v, bv, accO[t]);
    }
  }

  // ---- epilogue: normalize and add q residual ----
#pragma unroll
  for (int r = 0; r < 8; ++r) {
    float inv = 1.0f / lrun[r];
#pragma unroll
    for (int t = 0; t < 4; ++t) {
      int m = rowQ + rowHalf + r;
      int n = h * 64 + (t << 4) + colLane;
      float val = accO[t][r] * inv;
      outAttH[(size_t)m * DM + n] = val + qres[(size_t)m * DM + n];
    }
  }
}

// ---------------------------------------------------------------------------
// LayerNorm over D_MODEL=1024; writes f32 (for final residual) and bf16 (FFN A)
// ---------------------------------------------------------------------------
__global__ __launch_bounds__(256) void ln_kernel(
    const float* __restrict__ x, const float* __restrict__ g,
    const float* __restrict__ bb, float* __restrict__ yF, __bf16* __restrict__ yB) {
  const int DM = 1024;
  int row = blockIdx.x;
  const float* xr = x + (size_t)row * DM;
  __shared__ float red[256];
  int tid = threadIdx.x;

  float s = 0.0f;
  for (int i = tid; i < DM; i += 256) s += xr[i];
  red[tid] = s; __syncthreads();
  for (int st = 128; st > 0; st >>= 1) { if (tid < st) red[tid] += red[tid + st]; __syncthreads(); }
  float mu = red[0] * (1.0f / DM);
  __syncthreads();

  float v = 0.0f;
  for (int i = tid; i < DM; i += 256) { float d = xr[i] - mu; v += d * d; }
  red[tid] = v; __syncthreads();
  for (int st = 128; st > 0; st >>= 1) { if (tid < st) red[tid] += red[tid + st]; __syncthreads(); }
  float rs = rsqrtf(red[0] * (1.0f / DM) + 1e-5f);

  for (int i = tid; i < DM; i += 256) {
    float val = (xr[i] - mu) * rs * g[i] + bb[i];
    yF[(size_t)row * DM + i] = val;
    yB[(size_t)row * DM + i] = (__bf16)val;
  }
}

// ---------------------------------------------------------------------------
// Host launcher
// ---------------------------------------------------------------------------
extern "C" void kernel_launch(void* const* d_in, const int* in_sizes, int n_in,
                              void* d_out, int out_size, void* d_ws, size_t ws_size,
                              hipStream_t stream) {
  (void)in_sizes; (void)n_in; (void)out_size; (void)ws_size;
  const int S = 2048, DM = 1024, DFF = 4096;
  const int M = 2 * S;  // 4096 rows

  const float* q   = (const float*)d_in[0];
  const float* k   = (const float*)d_in[1];
  const float* Wq  = (const float*)d_in[2];  const float* bq = (const float*)d_in[3];
  const float* Wk  = (const float*)d_in[4];  const float* bk = (const float*)d_in[5];
  const float* Wv  = (const float*)d_in[6];  const float* bv = (const float*)d_in[7];
  const float* W1  = (const float*)d_in[8];  const float* b1 = (const float*)d_in[9];
  const float* W2  = (const float*)d_in[10]; const float* b2 = (const float*)d_in[11];
  const float* lng = (const float*)d_in[12]; const float* lnb = (const float*)d_in[13];
  const int*   causal = (const int*)d_in[14];
  float* out = (float*)d_out;

  char* ws = (char*)d_ws;
  size_t off = 0;
  auto alloc = [&](size_t bytes) -> char* {
    char* p = ws + off;
    off += (bytes + 255) & ~(size_t)255;
    return p;
  };

  __bf16* qb  = (__bf16*)alloc((size_t)M * DM * 2);
  __bf16* kb  = (__bf16*)alloc((size_t)M * DM * 2);
  __bf16* wqb = (__bf16*)alloc((size_t)DM * DM * 2);
  __bf16* wkb = (__bf16*)alloc((size_t)DM * DM * 2);
  __bf16* wvb = (__bf16*)alloc((size_t)DM * DM * 2);
  __bf16* w1b = (__bf16*)alloc((size_t)DFF * DM * 2);
  __bf16* w2b = (__bf16*)alloc((size_t)DM * DFF * 2);
  __bf16* Qb  = (__bf16*)alloc((size_t)M * DM * 2);
  __bf16* Kb  = (__bf16*)alloc((size_t)M * DM * 2);
  __bf16* Vt  = (__bf16*)alloc((size_t)M * DM * 2);   // head-transposed V
  float*  attH = (float*)alloc((size_t)M * DM * 4);   // att + q residual
  float*  lnF  = (float*)alloc((size_t)M * DM * 4);   // layernorm f32
  __bf16* lnB  = (__bf16*)alloc((size_t)M * DM * 2);  // layernorm bf16
  __bf16* Hb   = (__bf16*)alloc((size_t)M * DFF * 2); // gelu(ffn1) bf16

  auto castN = [&](const float* src, __bf16* dst, int n) {
    cast_f32_bf16<<<(n + 255) / 256, 256, 0, stream>>>(src, dst, n);
  };
  castN(q,  qb,  M * DM);
  castN(k,  kb,  M * DM);
  castN(Wq, wqb, DM * DM);
  castN(Wk, wkb, DM * DM);
  castN(Wv, wvb, DM * DM);
  castN(W1, w1b, DFF * DM);
  castN(W2, w2b, DM * DFF);

  // 8 waves / block; each wave does a 32x64 tile
  auto gemmBlocks = [&](int Mv, int Nv) { return ((Mv >> 5) * (Nv >> 6) + 7) / 8; };

  // Q/K/V projections
  gemm_wmma<MODE_QPROJ><<<gemmBlocks(M, DM), 256, 0, stream>>>(qb, wqb, bq, nullptr, nullptr, Qb, M, DM, DM);
  gemm_wmma<MODE_KPROJ><<<gemmBlocks(M, DM), 256, 0, stream>>>(kb, wkb, bk, nullptr, nullptr, Kb, M, DM, DM);
  gemm_wmma<MODE_VPROJ><<<gemmBlocks(M, DM), 256, 0, stream>>>(kb, wvb, bv, nullptr, nullptr, Vt, M, DM, DM);

  // flash attention: 2*16*128 = 4096 waves -> 512 blocks
  attn_flash<<<512, 256, 0, stream>>>(Qb, Kb, Vt, q, attH, causal);

  // layernorm
  ln_kernel<<<M, 256, 0, stream>>>(attH, lng, lnb, lnF, lnB);

  // FFN
  gemm_wmma<MODE_FF1><<<gemmBlocks(M, DFF), 256, 0, stream>>>(lnB, w1b, b1, nullptr, nullptr, Hb, M, DFF, DM);
  gemm_wmma<MODE_FF2><<<gemmBlocks(M, DM), 256, 0, stream>>>(Hb, w2b, b2, lnF, out, nullptr, M, DM, DFF);
}
